// LMModel_33758442946608
// MI455X (gfx1250) — compile-verified, matrix operand order checked by reading, not run on hardware
//
#include <hip/hip_runtime.h>
#include <hip/hip_bf16.h>

typedef __attribute__((ext_vector_type(16))) _Float16 v16h;
typedef __attribute__((ext_vector_type(8)))  _Float16 v8h;
typedef __attribute__((ext_vector_type(8)))  float    v8f;

union V16U { v16h v; v8h h[2]; };

// ---------------------------------------------------------------------------
// fp32 -> fp16 weight conversion
// ---------------------------------------------------------------------------
__global__ void f32_to_f16_kernel(const float* __restrict__ src,
                                  _Float16* __restrict__ dst, int n) {
    int i = blockIdx.x * blockDim.x + threadIdx.x;
    if (i < n) dst[i] = (_Float16)src[i];
}

// ---------------------------------------------------------------------------
// Embedding: out[r,h] = w_emb[h, src[r]] + b_emb[h], stored f16 row-major [BS,H]
// ---------------------------------------------------------------------------
__global__ void embed_kernel(const int* __restrict__ src,
                             const float* __restrict__ wemb,
                             const float* __restrict__ bemb,
                             _Float16* __restrict__ out,
                             int H, int V) {
    int r = blockIdx.x;          // row = b*S + s
    int h = threadIdx.x;         // H == blockDim.x == 512
    int v = src[r];
    out[(size_t)r * H + h] = (_Float16)(wemb[(size_t)h * V + v] + bemb[h]);
}

// ---------------------------------------------------------------------------
// WMMA fragment load helper (16-bit A/B layout:
// lanes 0-15 hold K 0-7 / 16-23, lanes 16-31 hold K 8-15 / 24-31;
// both halves are contiguous 16-byte runs -> two b128 loads)
// ---------------------------------------------------------------------------
__device__ __forceinline__ v16h load_frag(const _Float16* p) {
    V16U u;
    u.h[0] = *reinterpret_cast<const v8h*>(p);
    u.h[1] = *reinterpret_cast<const v8h*>(p + 16);
    return u.v;
}

// ---------------------------------------------------------------------------
// Generic WMMA GEMM: out[M,N] = A[M,K](f16) * W[N,K](f16)^T + bias0 + bias1
// 8 waves / 256-thread block. Each wave owns TPW consecutive 16-wide N-tiles.
// Per K-chunk: load the A fragment once and ALL TPW B fragments into distinct
// registers first, then issue the TPW WMMAs -> loads overlap compute instead
// of serializing on a single register bundle.
// ---------------------------------------------------------------------------
template <int TPW>
__global__ void gemm_wmma_kernel(const _Float16* __restrict__ A,
                                 const _Float16* __restrict__ W,
                                 const float* __restrict__ bias0,
                                 const float* __restrict__ bias1,
                                 float* __restrict__ outF,
                                 _Float16* __restrict__ outH,
                                 int M, int N, int K, int ldc, int relu) {
    const int lane  = threadIdx.x & 31;
    const int wave  = threadIdx.x >> 5;
    const int laneN = lane & 15;
    const int hi    = (lane >> 4) & 1;   // 0: K 0-7/16-23, 1: K 8-15/24-31

    const int nBase0 = (blockIdx.x * 8 + wave) * (16 * TPW);
    const int mBase  = blockIdx.y * 16;
    if (nBase0 >= N) return;             // wave-uniform: EXEC stays all-ones

    const _Float16* aRow = A + (size_t)(mBase + laneN) * K + hi * 8;
    const _Float16* wRow[TPW];
    #pragma unroll
    for (int t = 0; t < TPW; ++t)
        wRow[t] = W + (size_t)(nBase0 + t * 16 + laneN) * K + hi * 8;

    v8f acc[TPW];
    #pragma unroll
    for (int t = 0; t < TPW; ++t) acc[t] = (v8f){};

    for (int kc = 0; kc < K; kc += 32) {
        v16h bf[TPW];
        #pragma unroll
        for (int t = 0; t < TPW; ++t) bf[t] = load_frag(wRow[t] + kc);
        v16h af = load_frag(aRow + kc);
        #pragma unroll
        for (int t = 0; t < TPW; ++t)
            acc[t] = __builtin_amdgcn_wmma_f32_16x16x32_f16(
                        false, af, false, bf[t], (short)0, acc[t], false, false);
    }

    #pragma unroll
    for (int t = 0; t < TPW; ++t) {
        const int n = nBase0 + t * 16 + laneN;
        float bsum = 0.0f;
        if (bias0) bsum += bias0[n];
        if (bias1) bsum += bias1[n];
        #pragma unroll
        for (int i = 0; i < 8; ++i) {
            int m = mBase + hi * 8 + i;
            float vv = acc[t][i] + bsum;
            if (relu) vv = fmaxf(vv, 0.0f);
            if (outF) outF[(size_t)m * ldc + n] = vv;
            else      outH[(size_t)m * ldc + n] = (_Float16)vv;
        }
    }
}

// ---------------------------------------------------------------------------
// LSTM recurrence: single workgroup (32 waves, 1024 threads), sequential in S.
// h (f16) / c (f32) / gate staging (f16) in 112 KB of dynamic LDS.
// Each wave owns 64 gate columns (4 N-tiles, 4 accumulators); per K-chunk the
// h fragment is read from LDS once and all 4 w_hh fragments are loaded into
// distinct registers before the 4 WMMAs, hiding L2 load latency on the
// latency-critical sequential path. w_hh (f16, 2 MB) streams from L2.
//   gates[b, 0:2048] = xproj[b,s,:] + h @ w_hh^T
//   i,f,g,o split -> c = f*c + i*g ; h = o*tanh(c) ; h streamed to hseq.
// ---------------------------------------------------------------------------
__device__ __forceinline__ float sigmoidf(float x) {
    return 1.0f / (1.0f + __expf(-x));
}

__global__ void lstm_kernel(const float* __restrict__ xproj,   // [B,S,G] f32
                            const _Float16* __restrict__ whh,  // [G,H] f16
                            _Float16* __restrict__ hseq,       // [B,S,H] f16
                            int S, int H, int G) {
    extern __shared__ char smem[];
    _Float16* hbuf = reinterpret_cast<_Float16*>(smem);                 // [16][512]  16 KB
    float*    cbuf = reinterpret_cast<float*>(smem + 16 * 512 * 2);     // [16][512]  32 KB
    _Float16* gbuf = reinterpret_cast<_Float16*>(smem + 16 * 512 * 2
                                                      + 16 * 512 * 4); // [16][2048] 64 KB

    const int tid   = threadIdx.x;
    const int lane  = tid & 31;
    const int wave  = tid >> 5;
    const int laneN = lane & 15;
    const int hi    = (lane >> 4) & 1;

    // zero-init h and c
    for (int e = 0; e < 8; ++e) {
        int idx = e * 1024 + tid;       // 16*512 = 8192 elements
        hbuf[idx] = (_Float16)0.0f;
        cbuf[idx] = 0.0f;
    }
    __syncthreads();

    const int nBase0 = wave * 64;       // 32 waves x 64 cols = 2048 gate cols
    const _Float16* aRow = hbuf + laneN * H + hi * 8;   // batch row = laneN
    const _Float16* wRow[4];
    #pragma unroll
    for (int t = 0; t < 4; ++t)
        wRow[t] = whh + (size_t)(nBase0 + t * 16 + laneN) * H + hi * 8;

    for (int s = 0; s < S; ++s) {
        // ---- gates = h @ whh^T (+ xproj): 4 tiles per wave, shared A frag ----
        v8f acc[4];
        #pragma unroll
        for (int t = 0; t < 4; ++t) acc[t] = (v8f){};

        #pragma unroll 4
        for (int kc = 0; kc < H; kc += 32) {
            v16h bf[4];
            #pragma unroll
            for (int t = 0; t < 4; ++t) bf[t] = load_frag(wRow[t] + kc);
            V16U ua;
            ua.h[0] = *reinterpret_cast<const v8h*>(aRow + kc);        // ds_load_b128
            ua.h[1] = *reinterpret_cast<const v8h*>(aRow + kc + 16);
            #pragma unroll
            for (int t = 0; t < 4; ++t)
                acc[t] = __builtin_amdgcn_wmma_f32_16x16x32_f16(
                            false, ua.v, false, bf[t], (short)0, acc[t], false, false);
        }

        #pragma unroll
        for (int t = 0; t < 4; ++t) {
            const int col = nBase0 + t * 16 + laneN;
            #pragma unroll
            for (int i = 0; i < 8; ++i) {
                int b = hi * 8 + i;   // batch index (M == B == 16)
                float g = acc[t][i] + xproj[((size_t)b * S + s) * G + col];
                gbuf[b * G + col] = (_Float16)g;
            }
        }
        __syncthreads();

        // ---- elementwise LSTM cell: 8192 (=16*512) states, 8 per thread ----
        for (int e = 0; e < 8; ++e) {
            int idx = e * 1024 + tid;
            int b = idx >> 9;
            int j = idx & 511;
            float ig = sigmoidf((float)gbuf[b * G + j]);
            float fg = sigmoidf((float)gbuf[b * G + j + 512]);
            float gg = tanhf((float)gbuf[b * G + j + 1024]);
            float og = sigmoidf((float)gbuf[b * G + j + 1536]);
            float c  = fg * cbuf[idx] + ig * gg;
            cbuf[idx] = c;
            float h = og * tanhf(c);
            hbuf[idx] = (_Float16)h;
            hseq[((size_t)b * S + s) * H + j] = (_Float16)h;
        }
        __syncthreads();
    }
}

// ---------------------------------------------------------------------------
// Host orchestration
// ---------------------------------------------------------------------------
extern "C" void kernel_launch(void* const* d_in, const int* in_sizes, int n_in,
                              void* d_out, int out_size, void* d_ws, size_t ws_size,
                              hipStream_t stream) {
    (void)in_sizes; (void)n_in; (void)out_size; (void)ws_size;

    const int B = 16, S = 512, H = 512, V = 10000, G = 4 * H;
    const int BS = B * S;

    const int*   source = (const int*)  d_in[0];
    const float* w_emb  = (const float*)d_in[1];
    const float* b_emb  = (const float*)d_in[2];
    const float* w_ih0  = (const float*)d_in[3];
    const float* w_hh0  = (const float*)d_in[4];
    const float* b_ih0  = (const float*)d_in[5];
    const float* b_hh0  = (const float*)d_in[6];
    const float* w_ih1  = (const float*)d_in[7];
    const float* w_hh1  = (const float*)d_in[8];
    const float* b_ih1  = (const float*)d_in[9];
    const float* b_hh1  = (const float*)d_in[10];
    const float* w1     = (const float*)d_in[11];
    const float* b1     = (const float*)d_in[12];
    const float* w2     = (const float*)d_in[13];
    const float* b2     = (const float*)d_in[14];
    float* logits = (float*)d_out;

    // ---- workspace layout ----
    char* ws = (char*)d_ws;
    size_t off = 0;
    auto take = [&](size_t bytes) {
        char* p = ws + off;
        off = (off + bytes + 255) & ~size_t(255);
        return p;
    };
    _Float16* wih0h = (_Float16*)take(sizeof(_Float16) * (size_t)G * H);
    _Float16* whh0h = (_Float16*)take(sizeof(_Float16) * (size_t)G * H);
    _Float16* wih1h = (_Float16*)take(sizeof(_Float16) * (size_t)G * H);
    _Float16* whh1h = (_Float16*)take(sizeof(_Float16) * (size_t)G * H);
    _Float16* w1h   = (_Float16*)take(sizeof(_Float16) * (size_t)128 * H);
    _Float16* w2h   = (_Float16*)take(sizeof(_Float16) * (size_t)V * 128);
    _Float16* embh  = (_Float16*)take(sizeof(_Float16) * (size_t)BS * H);
    _Float16* hseq  = (_Float16*)take(sizeof(_Float16) * (size_t)BS * H);
    _Float16* hidh  = (_Float16*)take(sizeof(_Float16) * (size_t)BS * 128);
    float*    xproj = (float*)   take(sizeof(float)    * (size_t)BS * G);

    // ---- weight conversion fp32 -> fp16 ----
    auto conv = [&](const float* s, _Float16* d, int n) {
        f32_to_f16_kernel<<<(n + 255) / 256, 256, 0, stream>>>(s, d, n);
    };
    conv(w_ih0, wih0h, G * H);
    conv(w_hh0, whh0h, G * H);
    conv(w_ih1, wih1h, G * H);
    conv(w_hh1, whh1h, G * H);
    conv(w1,    w1h,   128 * H);
    conv(w2,    w2h,   V * 128);

    // ---- embedding gather ----
    embed_kernel<<<BS, H, 0, stream>>>(source, w_emb, b_emb, embh, H, V);

    const size_t lstm_lds = 16 * 512 * 2   // h f16
                          + 16 * 512 * 4   // c f32
                          + 16 * 2048 * 2; // gates f16  => 112 KB (< 320 KB/WGP)

    // ---- layer 0: input projection + recurrence ----
    gemm_wmma_kernel<4><<<dim3(G / 512, BS / 16), 256, 0, stream>>>(
        embh, wih0h, b_ih0, b_hh0, xproj, nullptr, BS, G, H, G, 0);
    lstm_kernel<<<1, 1024, lstm_lds, stream>>>(xproj, whh0h, hseq, S, H, G);

    // ---- layer 1 ----
    gemm_wmma_kernel<4><<<dim3(G / 512, BS / 16), 256, 0, stream>>>(
        hseq, wih1h, b_ih1, b_hh1, xproj, nullptr, BS, G, H, G, 0);
    lstm_kernel<<<1, 1024, lstm_lds, stream>>>(xproj, whh1h, hseq, S, H, G);

    // ---- hid = relu(h @ w1^T + b1): [8192,128] f16 ----
    gemm_wmma_kernel<1><<<dim3(1, BS / 16), 256, 0, stream>>>(
        hseq, w1h, b1, nullptr, nullptr, hidh, BS, 128, H, 128, 1);

    // ---- logits = hid @ w2^T + b2: [8192,10000] f32 -> d_out ----
    const int nTiles = V / 16;                 // 625 exact tiles
    gemm_wmma_kernel<1><<<dim3((nTiles + 7) / 8, BS / 16), 256, 0, stream>>>(
        hidh, w2h, b2, nullptr, logits, nullptr, BS, V, 128, V, 0);
}
